// TDTJoint_50723563766128
// MI455X (gfx1250) — matrix operand order repaired and checked by reading, not compiled, not run on hardware
//
#include <hip/hip_runtime.h>
#include <hip/hip_bf16.h>

// ---------------- CDNA5 WMMA types ----------------
typedef __attribute__((ext_vector_type(16))) _Float16 v16h;
typedef __attribute__((ext_vector_type(8)))  float    v8f;
typedef __attribute__((ext_vector_type(4)))  unsigned u32x4;
typedef __attribute__((ext_vector_type(4)))  float    f32x4;

union H16 {                // 16 halves = 32 bytes = 8 dwords
    v16h     f;            // WMMA fragment view
    u32x4    v[2];
    unsigned u[8];
    _Float16 h[16];
};
union HP { unsigned u; _Float16 h[2]; };

// Problem constants
#define ENC_  1024
#define PRED_ 640
#define H_    640
#define V_    1024
#define B_    4
#define T_    256
#define U_    64
#define M_    (B_*T_*U_)   // 65536
#define NDUR  5

// GEMM tiling
#define BM 128
#define BN 256
#define BK 32

__device__ __forceinline__ float fast_tanh(float x) {
    // tanh(x) = 1 - 2/(exp(2x)+1); saturates correctly at +-inf
    float e = __expf(2.0f * x);
    return 1.0f - 2.0f * __builtin_amdgcn_rcpf(e + 1.0f);
}

// A fragment (16x32 f16, MxK) from K-pair-packed LDS: ldsA[row*16 + kpair]
__device__ __forceinline__ v16h load_frag_a(const unsigned* ldsA, int row_base, int lane) {
    int r = row_base + (lane & 15);
    int g = (lane >> 4) & 1;
    const unsigned* p = ldsA + r * 16 + g * 4;
    H16 f;
#pragma unroll
    for (int v = 0; v < 4; ++v) f.u[v] = p[v];       // kpair = 4g + v   (K = 2v + 8g)
#pragma unroll
    for (int v = 4; v < 8; ++v) f.u[v] = p[v + 4];   // kpair = 8 + (v-4) + 4g (K = 16+2(v-4)+8g)
    return f.f;
}

// B fragment (32x16 f16, KxN) from K-pair-packed LDS: ldsB[kpair*BN + n]
__device__ __forceinline__ v16h load_frag_b(const unsigned* ldsB, int col_base, int lane) {
    int n = col_base + (lane & 15);
    int g = (lane >> 4) & 1;
    H16 f;
#pragma unroll
    for (int v = 0; v < 8; ++v) f.u[v] = ldsB[(g * 4 + v + (v >= 4 ? 4 : 0)) * BN + n]; // K = 8g + 2v (v<4), 16 + 8g + 2(v-4) (v>=4)
    return f.f;
}

// ---------------- prep: f32 row-projection -> f16 ----------------
// out[row, h] = (f16)( dot(X[row,:], W[h,:]) + bias[h] ),  row per block
__global__ __launch_bounds__(256) void proj_kernel(const float* __restrict__ X,
                                                   const float* __restrict__ W,
                                                   const float* __restrict__ bias,
                                                   _Float16* __restrict__ out, int K) {
    __shared__ float xr[1024];
    const int row = blockIdx.x;
    const float* xp = X + (size_t)row * K;
    for (int k = threadIdx.x; k < K; k += 256) xr[k] = xp[k];
    __syncthreads();
    for (int h = threadIdx.x; h < H_; h += 256) {
        const float* wp = W + (size_t)h * K;
        float acc = bias[h];
        for (int k = 0; k < K; k += 4) {
            f32x4 w4 = *(const f32x4*)(wp + k);
            acc += xr[k] * w4.x + xr[k + 1] * w4.y + xr[k + 2] * w4.z + xr[k + 3] * w4.w;
        }
        out[(size_t)row * H_ + h] = (_Float16)acc;
    }
}

// ---------------- prep: W_out [V,H] f32 -> Bmat [H,V] f16 ----------------
__global__ __launch_bounds__(256) void conv_wout(const float* __restrict__ Wout,
                                                 _Float16* __restrict__ Bm) {
    int idx = blockIdx.x * 256 + threadIdx.x;     // H_*V_ total
    int h = idx >> 10;                            // / V_
    int v = idx & (V_ - 1);
    Bm[idx] = (_Float16)Wout[(size_t)v * H_ + h];
}

// ---------------- main: logits = tanh(enc+pred) @ Bmat + b_out ----------------
__global__ __launch_bounds__(256) void joint_gemm(const _Float16* __restrict__ enc,
                                                  const _Float16* __restrict__ pred,
                                                  const _Float16* __restrict__ Bm,
                                                  const float* __restrict__ b_out,
                                                  float* __restrict__ out) {
    __shared__ unsigned ldsA[BM * (BK / 2)];   // 128 x 16 dwords = 8 KB  (K-pair packed)
    __shared__ unsigned ldsB[(BK / 2) * BN];   // 16 x 256 dwords = 16 KB (K-pair packed)

    const int tid  = threadIdx.x;
    const int lane = tid & 31;
    const int w    = tid >> 5;        // 8 waves
    const int wr   = w & 1;           // 2 row groups  (64 rows each)
    const int wc   = w >> 1;          // 4 col groups  (64 cols each)

    const int m0 = blockIdx.x * BM;   // gridDim.x = M_/BM = 512
    const int n0 = blockIdx.y * BN;   // gridDim.y = V_/BN = 4

    // A-fill indices: 2 threads per row, 16 halves each
    const int arow = tid >> 1;
    const int kq   = (tid & 1) << 4;             // 0 or 16 (half offset)
    const int m    = m0 + arow;
    const int bt   = m >> 6;                     // / U_
    const int uu   = m & (U_ - 1);
    const int prr  = ((m >> 14) << 6) | uu;      // (m / (T_*U_)) * U_ + u
    const _Float16* erow = enc  + (size_t)bt  * H_;
    const _Float16* prow = pred + (size_t)prr * H_;

    // B-fill indices: one K-pair, 16 columns per thread
    const int kp = tid >> 4;                     // 0..15
    const int cc = (tid & 15) << 4;              // 0..240

    const v8f zero = {0.f, 0.f, 0.f, 0.f, 0.f, 0.f, 0.f, 0.f};
    v8f acc[4][4];
#pragma unroll
    for (int i = 0; i < 4; ++i)
#pragma unroll
        for (int j = 0; j < 4; ++j) acc[i][j] = zero;

    for (int k0 = 0; k0 < H_; k0 += BK) {
        // ---- stage A tile: tanh(enc+pred) -> f16, K-pair packed ----
        {
            const u32x4* ep = (const u32x4*)(erow + k0 + kq);
            const u32x4* pp = (const u32x4*)(prow + k0 + kq);
            H16 E, P, A;
            E.v[0] = ep[0]; E.v[1] = ep[1];
            P.v[0] = pp[0]; P.v[1] = pp[1];
#pragma unroll
            for (int i = 0; i < 16; ++i)
                A.h[i] = (_Float16)fast_tanh((float)E.h[i] + (float)P.h[i]);
            u32x4* da = (u32x4*)&ldsA[arow * 16 + (kq >> 1)];
            da[0] = A.v[0]; da[1] = A.v[1];
        }
        // ---- stage B tile: two rows packed into K-pairs ----
        {
            const _Float16* b0 = Bm + (size_t)(k0 + 2 * kp) * V_ + n0 + cc;
            const u32x4* br0 = (const u32x4*)b0;
            const u32x4* br1 = (const u32x4*)(b0 + V_);
            H16 R0, R1;
            R0.v[0] = br0[0]; R0.v[1] = br0[1];
            R1.v[0] = br1[0]; R1.v[1] = br1[1];
            unsigned* db = &ldsB[kp * BN + cc];
#pragma unroll
            for (int i = 0; i < 16; ++i) {
                HP t; t.h[0] = R0.h[i]; t.h[1] = R1.h[i];
                db[i] = t.u;
            }
            if (k0 + BK < H_)  // prefetch next B tile (global_prefetch_b8)
                __builtin_prefetch((const void*)(b0 + (size_t)BK * V_), 0, 1);
        }
        __syncthreads();

        // ---- fragment loads + 16 WMMAs per wave ----
        v16h af[4], bf[4];
#pragma unroll
        for (int i = 0; i < 4; ++i) af[i] = load_frag_a(ldsA, wr * 64 + i * 16, lane);
#pragma unroll
        for (int j = 0; j < 4; ++j) bf[j] = load_frag_b(ldsB, wc * 64 + j * 16, lane);
#pragma unroll
        for (int i = 0; i < 4; ++i)
#pragma unroll
            for (int j = 0; j < 4; ++j)
                acc[i][j] = __builtin_amdgcn_wmma_f32_16x16x32_f16(
                    false, af[i], false, bf[j], (short)0, acc[i][j], false, false);
        __syncthreads();
    }

    // ---- epilogue: bias + store ----
    const int g  = lane >> 4;
    const int ln = lane & 15;
#pragma unroll
    for (int j = 0; j < 4; ++j) {
        const int col = n0 + wc * 64 + j * 16 + ln;
        const float bias = b_out[col];
#pragma unroll
        for (int i = 0; i < 4; ++i) {
            const int rbase = m0 + wr * 64 + i * 16 + g * 8;
#pragma unroll
            for (int v = 0; v < 8; ++v)
                out[(size_t)(rbase + v) * V_ + col] = acc[i][j][v] + bias;
        }
    }
}

// ---------------- durations head: one wave per (b,t,u) row ----------------
__global__ __launch_bounds__(256) void dur_kernel(const _Float16* __restrict__ enc,
                                                  const _Float16* __restrict__ pred,
                                                  const float* __restrict__ W_dur,
                                                  const float* __restrict__ b_dur,
                                                  float* __restrict__ out) {
    const int lane = threadIdx.x & 31;
    const int wave = threadIdx.x >> 5;
    const int m = blockIdx.x * 8 + wave;
    const int bt = m >> 6;
    const int uu = m & (U_ - 1);
    const int prr = ((m >> 14) << 6) | uu;
    const _Float16* ep = enc  + (size_t)bt  * H_;
    const _Float16* pp = pred + (size_t)prr * H_;
    float s[NDUR] = {0.f, 0.f, 0.f, 0.f, 0.f};
    for (int h = lane; h < H_; h += 32) {
        float a = fast_tanh((float)ep[h] + (float)pp[h]);
#pragma unroll
        for (int d = 0; d < NDUR; ++d) s[d] += a * W_dur[d * H_ + h];
    }
#pragma unroll
    for (int d = 0; d < NDUR; ++d)
#pragma unroll
        for (int off = 16; off > 0; off >>= 1)
            s[d] += __shfl_down(s[d], off, 32);
    if (lane == 0) {
#pragma unroll
        for (int d = 0; d < NDUR; ++d)
            out[(size_t)m * NDUR + d] = s[d] + b_dur[d];
    }
}

extern "C" void kernel_launch(void* const* d_in, const int* in_sizes, int n_in,
                              void* d_out, int out_size, void* d_ws, size_t ws_size,
                              hipStream_t stream) {
    (void)in_sizes; (void)n_in; (void)out_size; (void)ws_size;
    const float* encoder_out   = (const float*)d_in[0];
    const float* predictor_out = (const float*)d_in[1];
    const float* W_enc  = (const float*)d_in[2];
    const float* b_enc  = (const float*)d_in[3];
    const float* W_pred = (const float*)d_in[4];
    const float* b_pred = (const float*)d_in[5];
    const float* W_out  = (const float*)d_in[6];
    const float* b_out  = (const float*)d_in[7];
    const float* W_dur  = (const float*)d_in[8];
    const float* b_dur  = (const float*)d_in[9];

    // workspace carve-up (all 256B-aligned)
    char* ws = (char*)d_ws;
    _Float16* enc16  = (_Float16*)(ws);                      // 1024*640*2 = 1310720 B
    _Float16* pred16 = (_Float16*)(ws + 1310720);            //  256*640*2 =  327680 B
    _Float16* Bmat   = (_Float16*)(ws + 1310720 + 327680);   //  640*1024*2 = 1310720 B

    float* logits = (float*)d_out;                           // [B,T,U,V]
    float* durs   = logits + (size_t)M_ * V_;                // [B,T,U,5]

    proj_kernel<<<B_ * T_, 256, 0, stream>>>(encoder_out,   W_enc,  b_enc,  enc16,  ENC_);
    proj_kernel<<<B_ * U_, 256, 0, stream>>>(predictor_out, W_pred, b_pred, pred16, PRED_);
    conv_wout<<<(H_ * V_) / 256, 256, 0, stream>>>(W_out, Bmat);

    dim3 grid(M_ / BM, V_ / BN);   // (512, 4)
    joint_gemm<<<grid, 256, 0, stream>>>(enc16, pred16, Bmat, b_out, logits);

    dur_kernel<<<M_ / 8, 256, 0, stream>>>(enc16, pred16, W_dur, b_dur, durs);
}
